// TensorTrainComplexLinear_29351806501337
// MI455X (gfx1250) — matrix operand order repaired and checked by reading, not compiled, not run on hardware
//
#include <hip/hip_runtime.h>
#include <hip/hip_bf16.h>

// ---------------------------------------------------------------------------
// TensorTrainComplexLinear on MI455X (gfx1250)
//   out = [x_r x_i] @ [[Wr, -Wi],[Wi, Wr]]^T + bias   -> single 8192x4096x4096 GEMM
//   Precision: bf16 hi/lo split (3-term) on v_wmma_f32_16x16x32_bf16, f32 accum
//   Wave tile 32(M) x 128(N) = 2x8 WMMA tiles; B fragments double-buffered in
//   registers, A raw loads pipelined one K-step ahead.
// ---------------------------------------------------------------------------

typedef __attribute__((ext_vector_type(16))) __bf16 v16bf;
typedef __attribute__((ext_vector_type(8)))  float  v8f;
typedef __attribute__((ext_vector_type(4)))  float  v4f;

#define IN_F   2048
#define OUT_F  2048
#define KDIM   4096   // 2*IN_F
#define NDIM   4096   // 2*OUT_F
#define MDIM   8192   // 4*2048 rows
#define TT_RANK 4

// ---- Kernel 1: materialize Wcat as bf16 hi/lo, row-major [NDIM][KDIM] -----
__global__ __launch_bounds__(256)
void build_w(const float* __restrict__ G1R, const float* __restrict__ G2R,
             const float* __restrict__ G1I, const float* __restrict__ G2I,
             __bf16* __restrict__ Whi, __bf16* __restrict__ Wlo) {
    unsigned idx = blockIdx.x * 256u + threadIdx.x;   // over NDIM*KDIM = 16.7M
    unsigned k = idx & (KDIM - 1);
    unsigned j = idx >> 12;                            // /4096
    if (j >= NDIM) return;

    unsigned n  = j & (OUT_F - 1);
    unsigned kk = k & (IN_F - 1);
    bool jimag = j >= OUT_F;
    bool kimag = k >= IN_F;

    unsigned o1 = n >> 7, o2 = n & 127;
    unsigned i1 = kk >> 7, i2 = kk & 127;

    // quadrants: (r,r)->+Wr  (r,i)->-Wi  (i,r)->+Wi  (i,i)->+Wr
    const float* G1 = (jimag != kimag) ? G1I : G1R;
    const float* G2 = (jimag != kimag) ? G2I : G2R;

    float w = 0.f;
#pragma unroll
    for (int r = 0; r < TT_RANK; ++r)
        w += G1[o1 * (TT_RANK * 16) + r * 16 + i1] *
             G2[(unsigned)r * (128u * 128u) + o2 * 128u + i2];
    if (!jimag && kimag) w = -w;

    __bf16 hi = (__bf16)w;
    __bf16 lo = (__bf16)(w - (float)hi);
    Whi[idx] = hi;
    Wlo[idx] = lo;
}

// ---- Kernel 2: WMMA GEMM ---------------------------------------------------
// Block = 256 threads = 8 waves laid out 4(M) x 2(N). Wave tile = 32(M) x 128(N)
// = 2x8 grid of 16x16 WMMA tiles. Workgroup tile = 128 x 256.
__global__ __launch_bounds__(256)
void tt_gemm(const float*  __restrict__ X,     // [MDIM][KDIM] f32
             const __bf16* __restrict__ Whi,   // [NDIM][KDIM] bf16
             const __bf16* __restrict__ Wlo,   // [NDIM][KDIM] bf16
             const float*  __restrict__ bias_r,// [OUT_F]
             const float*  __restrict__ bias_i,// [OUT_F]
             float*        __restrict__ Out)   // [MDIM][NDIM] f32
{
    const int lane = threadIdx.x & 31;
    const int wave = threadIdx.x >> 5;
    const int wm   = wave & 3;                         // 0..3  (M direction)
    const int wn   = wave >> 2;                        // 0..1  (N direction)
    const int m0   = blockIdx.y * 128 + wm * 32;       // wave M base (2 tiles)
    const int n0   = blockIdx.x * 256 + wn * 128;      // wave N base (8 tiles)
    const int lr   = lane & 15;                        // row/col within tile
    const int half = lane >> 4;                        // lane half-group

    const v8f vzero = {0.f, 0.f, 0.f, 0.f, 0.f, 0.f, 0.f, 0.f};
    v8f acc[2][8];
#pragma unroll
    for (int im = 0; im < 2; ++im)
#pragma unroll
        for (int in = 0; in < 8; ++in) acc[im][in] = vzero;

    // A rows: lane lr reads row m0+lr (and m0+16+lr); half selects K slice
    const float* xrow[2];
    xrow[0] = X + (size_t)(m0 + lr) * KDIM + half * 16;
    xrow[1] = X + (size_t)(m0 + 16 + lr) * KDIM + half * 16;

    // B rows: lane lr reads W row n0 + 16*in + lr; half selects K slice
    const __bf16* wbase = Whi + (size_t)(n0 + lr) * KDIM + half * 16;
    const __bf16* lbase = Wlo + (size_t)(n0 + lr) * KDIM + half * 16;

    // --- A raw prefetch pipeline: araw holds f32 data for the CURRENT kb ---
    v4f araw[2][4];
#pragma unroll
    for (int im = 0; im < 2; ++im) {
        araw[im][0] = *(const v4f*)(xrow[im] + 0);
        araw[im][1] = *(const v4f*)(xrow[im] + 4);
        araw[im][2] = *(const v4f*)(xrow[im] + 8);
        araw[im][3] = *(const v4f*)(xrow[im] + 12);
    }

    // --- B double buffer: bcur holds hi/lo for the tile about to be used ---
    v16bf bh_cur = *(const v16bf*)(wbase);
    v16bf bl_cur = *(const v16bf*)(lbase);

    for (int kb = 0; kb < KDIM; kb += 32) {
        const int kbn = (kb + 32) & (KDIM - 1);        // wraps on last step (harmless)

        // Convert current A raw f32 -> bf16 hi/lo (pure VALU, co-executes w/ WMMA)
        //   A 16x32 layout (matches B):
        //   elem e : K = kb + half*16 + e
        v16bf Ahi[2], Alo[2];
#pragma unroll
        for (int im = 0; im < 2; ++im) {
            float av[16];
            *(v4f*)(av + 0)  = araw[im][0];
            *(v4f*)(av + 4)  = araw[im][1];
            *(v4f*)(av + 8)  = araw[im][2];
            *(v4f*)(av + 12) = araw[im][3];
#pragma unroll
            for (int e = 0; e < 16; ++e) {
                float  v = av[e];
                __bf16 h = (__bf16)v;
                Ahi[im][e] = h;
                Alo[im][e] = (__bf16)(v - (float)h);
            }
        }

        // Issue A raw loads for NEXT kb now -> a full K-step of latency cover
#pragma unroll
        for (int im = 0; im < 2; ++im) {
            araw[im][0] = *(const v4f*)(xrow[im] + kbn + 0);
            araw[im][1] = *(const v4f*)(xrow[im] + kbn + 4);
            araw[im][2] = *(const v4f*)(xrow[im] + kbn + 8);
            araw[im][3] = *(const v4f*)(xrow[im] + kbn + 12);
        }

        // --- N-tile loop: B software-pipelined one tile ahead ---
        //   B 32x16 order used here (same as A): elem e : K = kb + half*16 + e (contig 32B)
#pragma unroll
        for (int in = 0; in < 8; ++in) {
            v16bf bh_nxt, bl_nxt;
            if (in < 7) {
                const size_t roff = (size_t)(in + 1) * 16 * KDIM + kb;
                bh_nxt = *(const v16bf*)(wbase + roff);
                bl_nxt = *(const v16bf*)(lbase + roff);
            } else {
                bh_nxt = *(const v16bf*)(wbase + kbn);   // in=0 of next K-step
                bl_nxt = *(const v16bf*)(lbase + kbn);
            }

#pragma unroll
            for (int im = 0; im < 2; ++im) {
                acc[im][in] = __builtin_amdgcn_wmma_f32_16x16x32_bf16(
                    false, Ahi[im], false, bh_cur, (short)0, acc[im][in], false, false);
                acc[im][in] = __builtin_amdgcn_wmma_f32_16x16x32_bf16(
                    false, Alo[im], false, bh_cur, (short)0, acc[im][in], false, false);
                acc[im][in] = __builtin_amdgcn_wmma_f32_16x16x32_bf16(
                    false, Ahi[im], false, bl_cur, (short)0, acc[im][in], false, false);
            }
            bh_cur = bh_nxt;
            bl_cur = bl_nxt;
        }
    }

    // --- Epilogue: C/D 16x16 f32 layout: lane col N=lr, VGPR v -> M = v + 8*half
#pragma unroll
    for (int in = 0; in < 8; ++in) {
        int j = n0 + in * 16 + lr;
        float b = (j < OUT_F) ? bias_r[j] : bias_i[j - OUT_F];
#pragma unroll
        for (int im = 0; im < 2; ++im) {
#pragma unroll
            for (int v = 0; v < 8; ++v) {
                int m = m0 + im * 16 + v + half * 8;
                Out[(size_t)m * NDIM + j] = acc[im][in][v] + b;
            }
        }
    }
}

// ---------------------------------------------------------------------------
extern "C" void kernel_launch(void* const* d_in, const int* in_sizes, int n_in,
                              void* d_out, int out_size, void* d_ws, size_t ws_size,
                              hipStream_t stream) {
    (void)in_sizes; (void)n_in; (void)out_size; (void)ws_size;
    const float* x   = (const float*)d_in[0];
    const float* G1R = (const float*)d_in[1];
    const float* G2R = (const float*)d_in[2];
    const float* G1I = (const float*)d_in[3];
    const float* G2I = (const float*)d_in[4];
    const float* br  = (const float*)d_in[5];
    const float* bi  = (const float*)d_in[6];

    __bf16* Whi = (__bf16*)d_ws;                       // 33.55 MB
    __bf16* Wlo = Whi + (size_t)NDIM * KDIM;           // 33.55 MB

    build_w<<<(NDIM * KDIM) / 256, 256, 0, stream>>>(G1R, G2R, G1I, G2I, Whi, Wlo);

    dim3 grid(NDIM / 256, MDIM / 128);                 // 16 x 64 workgroups
    tt_gemm<<<grid, 256, 0, stream>>>(x, Whi, Wlo, br, bi, (float*)d_out);
}
